// MoELayerParallel_7499012899151
// MI455X (gfx1250) — matrix operand order repaired and checked by reading, not compiled, not run on hardware
//
#include <hip/hip_runtime.h>

// Problem constants (reference: B=2,S=2048,D=1024,E=8,H=4096,k=2)
#define T_TOK 4096
#define DDIM  1024
#define EEXP  8
#define HDIM  4096
#define ROWS_CAP 8704          // k*T (=8192) + per-expert padding to 64 (<=8*63)
#define MAX_TILES 136          // ROWS_CAP/64
#define BM 64
#define BN 128
#define BK 32
#define LDA 40                 // bf16 elements per LDS row (80B: 16B-aligned frags, conflict-free)
#define LDB 40

typedef __bf16 bf16;
typedef bf16 v16bf __attribute__((ext_vector_type(16)));
typedef bf16 v8bf  __attribute__((ext_vector_type(8)));
typedef float v8f  __attribute__((ext_vector_type(8)));

__device__ __forceinline__ v16bf frag_cat(v8bf lo, v8bf hi) {
  v16bf r;
#pragma unroll
  for (int i = 0; i < 8; ++i) { r[i] = lo[i]; r[i + 8] = hi[i]; }
  return r;
}

// CDNA5 async global->LDS copy, 16B per lane (tracked by ASYNCcnt).
__device__ __forceinline__ void async_b128(unsigned lds_off, const void* g) {
  asm volatile("global_load_async_to_lds_b128 %0, %1, off"
               :: "v"(lds_off), "v"((unsigned long long)(uintptr_t)g)
               : "memory");
}
__device__ __forceinline__ void wait_async0() {
  asm volatile("s_wait_asynccnt 0" ::: "memory");
}
// Low 32 bits of a generic LDS pointer == wave-relative LDS byte offset.
__device__ __forceinline__ unsigned lds_off(const void* p) {
  return (unsigned)(uintptr_t)p;
}

// ---------------- init: zero output + control state (atomics accumulate -> every call)
__global__ void moe_init(float* out, int* rowtok, int* tile_expert,
                         int* counts, int* fill, float* usage, bf16* zrow) {
  int i = blockIdx.x * blockDim.x + threadIdx.x;
  if (i < T_TOK * DDIM) out[i] = 0.f;
  if (i < ROWS_CAP) rowtok[i] = -1;
  if (i < MAX_TILES) tile_expert[i] = -1;
  if (i < EEXP) { counts[i] = 0; fill[i] = 0; usage[i] = 0.f; }
  if (i < 32) zrow[i] = (bf16)0.f;
}

// ---------------- one-shot fp32 -> bf16 conversion (8 elems/thread)
__global__ __launch_bounds__(256) void moe_cvt(const float* __restrict__ src,
                                               bf16* __restrict__ dst, int n8) {
  int i = blockIdx.x * 256 + threadIdx.x;
  if (i >= n8) return;
  const float4* s = (const float4*)src + (size_t)i * 2;
  float4 a = s[0], b = s[1];
  bf16* d = dst + (size_t)i * 8;
  d[0] = (bf16)a.x; d[1] = (bf16)a.y; d[2] = (bf16)a.z; d[3] = (bf16)a.w;
  d[4] = (bf16)b.x; d[5] = (bf16)b.y; d[6] = (bf16)b.z; d[7] = (bf16)b.w;
}

// ---------------- gating logits (fp32 path: routing decisions stay exact)
__global__ __launch_bounds__(256) void moe_gate(const float* __restrict__ x,
                                                const float* __restrict__ gw,
                                                float* __restrict__ logits) {
  int wid = threadIdx.x >> 5, lane = threadIdx.x & 31;
  int t = blockIdx.x * 8 + wid;
  const float* xr = x + (size_t)t * DDIM;
  float acc[EEXP];
#pragma unroll
  for (int e = 0; e < EEXP; ++e) acc[e] = 0.f;
  for (int d = lane; d < DDIM; d += 32) {
    float xv = xr[d];
#pragma unroll
    for (int e = 0; e < EEXP; ++e) acc[e] += xv * gw[e * DDIM + d];
  }
#pragma unroll
  for (int e = 0; e < EEXP; ++e) {
#pragma unroll
    for (int off = 16; off; off >>= 1) acc[e] += __shfl_xor(acc[e], off, 32);
  }
  if (lane == 0) {
#pragma unroll
    for (int e = 0; e < EEXP; ++e) logits[t * EEXP + e] = acc[e];
  }
}

// ---------------- routing pass 1
__global__ __launch_bounds__(256) void moe_route1(const float* __restrict__ logits,
                                                  const float* __restrict__ noise,
                                                  const float* __restrict__ nw,
                                                  int* __restrict__ top_idx,
                                                  float* __restrict__ top_w,
                                                  int* counts, float* usage) {
  __shared__ float su[EEXP];
  __shared__ int sc[EEXP];
  int tid = threadIdx.x;
  if (tid < EEXP) { su[tid] = 0.f; sc[tid] = 0; }
  __syncthreads();
  int t = blockIdx.x * blockDim.x + tid;
  float l[EEXP], ln[EEXP], p[EEXP];
  float mx = -1e30f;
#pragma unroll
  for (int e = 0; e < EEXP; ++e) { l[e] = logits[t * EEXP + e]; mx = fmaxf(mx, l[e]); }
  float s = 0.f;
#pragma unroll
  for (int e = 0; e < EEXP; ++e) { p[e] = __expf(l[e] - mx); s += p[e]; }
  float inv = 1.f / s;
#pragma unroll
  for (int e = 0; e < EEXP; ++e) { p[e] *= inv; atomicAdd(&su[e], p[e]); }
#pragma unroll
  for (int e = 0; e < EEXP; ++e) ln[e] = l[e] + noise[t * EEXP + e] * nw[e];  // NOISY_STD=1
  int i1 = 0;
#pragma unroll
  for (int e = 1; e < EEXP; ++e) if (ln[e] > ln[i1]) i1 = e;
  int i2 = (i1 == 0) ? 1 : 0;
#pragma unroll
  for (int e = 0; e < EEXP; ++e) if (e != i1 && ln[e] > ln[i2]) i2 = e;
  float e2 = __expf(ln[i2] - ln[i1]);
  float d = 1.f / (1.f + e2);
  top_idx[t * 2] = i1; top_idx[t * 2 + 1] = i2;
  top_w[t * 2] = d;    top_w[t * 2 + 1] = e2 * d;
  atomicAdd(&sc[i1], 1); atomicAdd(&sc[i2], 1);
  __syncthreads();
  if (tid < EEXP) { atomicAdd(&counts[tid], sc[tid]); atomicAdd(&usage[tid], su[tid]); }
}

// ---------------- setup: padded per-expert offsets, tile->expert map, lb_loss
__global__ void moe_setup(const int* counts, const float* usage, int* offsets,
                          int* tile_expert, float* out_loss) {
  if (threadIdx.x == 0 && blockIdx.x == 0) {
    int off = 0;
    for (int e = 0; e < EEXP; ++e) {
      offsets[e] = off;
      int tiles = (counts[e] + BM - 1) / BM;
      for (int tt = 0; tt < tiles; ++tt) tile_expert[off / BM + tt] = e;
      off += tiles * BM;
    }
    float acc = 0.f;
    for (int e = 0; e < EEXP; ++e) {
      float u = usage[e] / (float)T_TOK - 1.0f / EEXP;
      acc += u * u;
    }
    out_loss[0] = (acc / EEXP) * 0.01f;  // LOAD_BALANCE_SCALE
  }
}

// ---------------- routing pass 2: compact (token, weight) rows per expert
__global__ __launch_bounds__(256) void moe_route2(const int* __restrict__ top_idx,
                                                  const float* __restrict__ top_w,
                                                  const int* __restrict__ offsets,
                                                  int* fill, int* rowtok, float* rowwgt) {
  int t = blockIdx.x * blockDim.x + threadIdx.x;
#pragma unroll
  for (int j = 0; j < 2; ++j) {
    int e = top_idx[t * 2 + j];
    int pos = atomicAdd(&fill[e], 1);
    int row = offsets[e] + pos;
    rowtok[row] = t;
    rowwgt[row] = top_w[t * 2 + j];
  }
}

// ---------------- FFN stage 1: h = (x@w1^T + b1) * silu(x@w2^T + b2) -> bf16 h_act
// Async double-buffered pipeline: one barrier per K-step, loads overlap WMMA.
__global__ __launch_bounds__(256) void moe_ffn1(
    const bf16* __restrict__ xbf, const bf16* __restrict__ w1, const float* __restrict__ b1,
    const bf16* __restrict__ w2, const float* __restrict__ b2,
    const int* __restrict__ tile_expert, const int* __restrict__ rowtok,
    const bf16* __restrict__ zrow, bf16* __restrict__ hact) {
  const int te = tile_expert[blockIdx.x];
  if (te < 0) return;                      // block-uniform: EXEC all-ones at WMMAs
  const int rowBase = blockIdx.x * BM;
  const int nBase = blockIdx.y * BN;
  __shared__ bf16 sA[2][BM * LDA];
  __shared__ bf16 sB1[2][BN * LDB];
  __shared__ bf16 sB2[2][BN * LDB];
  const int tid = threadIdx.x, lane = tid & 31, wid = tid >> 5;
  const int wm = (wid & 3) * 16;
  const int wn = (wid >> 2) * 64;

  // per-thread async copy tasks (all 16B, 16B-aligned in LDS thanks to LDA/LDB=40)
  const int ar = tid >> 2, ac8 = (tid & 3) * 8;          // A: 64 rows x 4 chunks
  const int tok = rowtok[rowBase + ar];
  const bf16* aSrcRow = (tok >= 0) ? xbf + (size_t)tok * DDIM + ac8 : zrow;
  const int br0 = tid >> 1, bc0 = (tid & 1) * 8;         // B: tasks tid, tid+256
  const int br1 = (tid + 256) >> 1, bc1 = ((tid + 256) & 1) * 8;
  const size_t wbase = ((size_t)te * HDIM + nBase) * DDIM;

  v8f acc1[4], acc2[4];
  v8f zero;
#pragma unroll
  for (int i = 0; i < 8; ++i) zero[i] = 0.f;
#pragma unroll
  for (int f = 0; f < 4; ++f) { acc1[f] = zero; acc2[f] = zero; }

  auto issue = [&](int k0, int buf) {
    async_b128(lds_off(&sA[buf][ar * LDA + ac8]), (tok >= 0) ? aSrcRow + k0 : zrow);
    async_b128(lds_off(&sB1[buf][br0 * LDB + bc0]), w1 + wbase + (size_t)br0 * DDIM + k0 + bc0);
    async_b128(lds_off(&sB2[buf][br0 * LDB + bc0]), w2 + wbase + (size_t)br0 * DDIM + k0 + bc0);
    async_b128(lds_off(&sB1[buf][br1 * LDB + bc1]), w1 + wbase + (size_t)br1 * DDIM + k0 + bc1);
    async_b128(lds_off(&sB2[buf][br1 * LDB + bc1]), w2 + wbase + (size_t)br1 * DDIM + k0 + bc1);
  };

  issue(0, 0);
  int buf = 0;
  for (int k0 = 0; k0 < DDIM; k0 += BK, buf ^= 1) {
    wait_async0();          // my tile-k writes landed
    __syncthreads();        // everyone's writes landed; everyone's reads of buf^1 done
    if (k0 + BK < DDIM) issue(k0 + BK, buf ^ 1);  // overlaps with WMMA below
    const int m = wm + (lane & 15);
    const int kb = (lane < 16) ? 0 : 16;
    v16bf a = frag_cat(*(const v8bf*)&sA[buf][m * LDA + kb],
                       *(const v8bf*)&sA[buf][m * LDA + kb + 8]);
    const int bb = (lane < 16) ? 0 : 16;
#pragma unroll
    for (int f = 0; f < 4; ++f) {
      int n = wn + f * 16 + (lane & 15);
      v16bf bA = frag_cat(*(const v8bf*)&sB1[buf][n * LDB + bb],
                          *(const v8bf*)&sB1[buf][n * LDB + bb + 8]);
      acc1[f] = __builtin_amdgcn_wmma_f32_16x16x32_bf16(false, a, false, bA,
                                                        (short)0, acc1[f], false, false);
      v16bf bB = frag_cat(*(const v8bf*)&sB2[buf][n * LDB + bb],
                          *(const v8bf*)&sB2[buf][n * LDB + bb + 8]);
      acc2[f] = __builtin_amdgcn_wmma_f32_16x16x32_bf16(false, a, false, bB,
                                                        (short)0, acc2[f], false, false);
    }
  }
  // epilogue: SwiGLU -> bf16 h_act
  int mhalf = (lane >> 4) * 8;
#pragma unroll
  for (int f = 0; f < 4; ++f) {
    int col = nBase + wn + f * 16 + (lane & 15);
    float bb1 = b1[te * HDIM + col];
    float bb2 = b2[te * HDIM + col];
#pragma unroll
    for (int i = 0; i < 8; ++i) {
      int row = rowBase + wm + i + mhalf;
      float g1 = acc1[f][i] + bb1;
      float g2 = acc2[f][i] + bb2;
      float h = g1 * g2 * (1.f / (1.f + __expf(-g2)));
      hact[(size_t)row * HDIM + col] = (bf16)h;
    }
  }
}

// ---------------- FFN stage 2: out[tok] += w * (h_act @ wp^T + bp)
__global__ __launch_bounds__(256) void moe_ffn2(
    const bf16* __restrict__ hact, const bf16* __restrict__ wp, const float* __restrict__ bp,
    const int* __restrict__ tile_expert, const int* __restrict__ rowtok,
    const float* __restrict__ rowwgt, float* __restrict__ out) {
  const int te = tile_expert[blockIdx.x];
  if (te < 0) return;
  const int rowBase = blockIdx.x * BM;
  const int nBase = blockIdx.y * BN;       // over D
  __shared__ bf16 sA[2][BM * LDA];
  __shared__ bf16 sB[2][BN * LDB];
  const int tid = threadIdx.x, lane = tid & 31, wid = tid >> 5;
  const int wm = (wid & 3) * 16;
  const int wn = (wid >> 2) * 64;

  const int ar = tid >> 2, ac8 = (tid & 3) * 8;
  const bf16* aSrcRow = hact + (size_t)(rowBase + ar) * HDIM + ac8;
  const int br0 = tid >> 1, bc0 = (tid & 1) * 8;
  const int br1 = (tid + 256) >> 1, bc1 = ((tid + 256) & 1) * 8;
  const size_t wbase = ((size_t)te * DDIM + nBase) * HDIM;

  v8f acc[4];
  v8f zero;
#pragma unroll
  for (int i = 0; i < 8; ++i) zero[i] = 0.f;
#pragma unroll
  for (int f = 0; f < 4; ++f) acc[f] = zero;

  auto issue = [&](int k0, int buf) {
    async_b128(lds_off(&sA[buf][ar * LDA + ac8]), aSrcRow + k0);
    async_b128(lds_off(&sB[buf][br0 * LDB + bc0]), wp + wbase + (size_t)br0 * HDIM + k0 + bc0);
    async_b128(lds_off(&sB[buf][br1 * LDB + bc1]), wp + wbase + (size_t)br1 * HDIM + k0 + bc1);
  };

  issue(0, 0);
  int buf = 0;
  for (int k0 = 0; k0 < HDIM; k0 += BK, buf ^= 1) {
    wait_async0();
    __syncthreads();
    if (k0 + BK < HDIM) issue(k0 + BK, buf ^ 1);
    const int m = wm + (lane & 15);
    const int kb = (lane < 16) ? 0 : 16;
    v16bf a = frag_cat(*(const v8bf*)&sA[buf][m * LDA + kb],
                       *(const v8bf*)&sA[buf][m * LDA + kb + 8]);
    const int bb = (lane < 16) ? 0 : 16;
#pragma unroll
    for (int f = 0; f < 4; ++f) {
      int n = wn + f * 16 + (lane & 15);
      v16bf b = frag_cat(*(const v8bf*)&sB[buf][n * LDB + bb],
                         *(const v8bf*)&sB[buf][n * LDB + bb + 8]);
      acc[f] = __builtin_amdgcn_wmma_f32_16x16x32_bf16(false, a, false, b,
                                                       (short)0, acc[f], false, false);
    }
  }
  int mhalf = (lane >> 4) * 8;
#pragma unroll
  for (int f = 0; f < 4; ++f) {
    int col = nBase + wn + f * 16 + (lane & 15);
    float bpv = bp[te * DDIM + col];
#pragma unroll
    for (int i = 0; i < 8; ++i) {
      int row = rowBase + wm + i + mhalf;
      int tok = rowtok[row];
      if (tok >= 0) {
        float wgt = rowwgt[row];
        unsafeAtomicAdd(out + (size_t)tok * DDIM + col, wgt * (acc[f][i] + bpv));
      }
    }
  }
}

extern "C" void kernel_launch(void* const* d_in, const int* in_sizes, int n_in,
                              void* d_out, int out_size, void* d_ws, size_t ws_size,
                              hipStream_t stream) {
  (void)in_sizes; (void)n_in; (void)out_size; (void)ws_size;
  const float* x     = (const float*)d_in[0];
  const float* noise = (const float*)d_in[1];
  // d_in[2] = k == 2 (fixed by reference shapes; kernels hardcode top-2)
  const float* gw = (const float*)d_in[3];
  const float* nw = (const float*)d_in[4];
  const float* w1 = (const float*)d_in[5];
  const float* b1 = (const float*)d_in[6];
  const float* w2 = (const float*)d_in[7];
  const float* b2 = (const float*)d_in[8];
  const float* wp = (const float*)d_in[9];
  const float* bp = (const float*)d_in[10];
  float* out = (float*)d_out;
  float* out_loss = out + (size_t)T_TOK * DDIM;

  char* p = (char*)d_ws;
  auto alloc = [&](size_t bytes) -> char* {
    char* r = p; p += (bytes + 255) & ~(size_t)255; return r;
  };
  float* logits  = (float*)alloc(sizeof(float) * T_TOK * EEXP);
  int* top_idx   = (int*)alloc(sizeof(int) * T_TOK * 2);
  float* top_w   = (float*)alloc(sizeof(float) * T_TOK * 2);
  int* counts    = (int*)alloc(sizeof(int) * EEXP);
  int* fill      = (int*)alloc(sizeof(int) * EEXP);
  float* usage   = (float*)alloc(sizeof(float) * EEXP);
  int* offsets   = (int*)alloc(sizeof(int) * EEXP);
  int* tile_exp  = (int*)alloc(sizeof(int) * MAX_TILES);
  int* rowtok    = (int*)alloc(sizeof(int) * ROWS_CAP);
  float* rowwgt  = (float*)alloc(sizeof(float) * ROWS_CAP);
  bf16* zrow     = (bf16*)alloc(sizeof(bf16) * 32);
  bf16* xbf      = (bf16*)alloc(sizeof(bf16) * (size_t)T_TOK * DDIM);
  bf16* w1bf     = (bf16*)alloc(sizeof(bf16) * (size_t)EEXP * HDIM * DDIM);
  bf16* w2bf     = (bf16*)alloc(sizeof(bf16) * (size_t)EEXP * HDIM * DDIM);
  bf16* wpbf     = (bf16*)alloc(sizeof(bf16) * (size_t)EEXP * DDIM * HDIM);
  bf16* hact     = (bf16*)alloc(sizeof(bf16) * (size_t)ROWS_CAP * HDIM);  // ~71 MB

  const int NW8 = (EEXP * HDIM * DDIM) / 8;   // weight elements / 8
  const int NX8 = (T_TOK * DDIM) / 8;

  moe_init<<<(T_TOK * DDIM + 255) / 256, 256, 0, stream>>>(out, rowtok, tile_exp,
                                                           counts, fill, usage, zrow);
  moe_cvt<<<(NX8 + 255) / 256, 256, 0, stream>>>(x, xbf, NX8);
  moe_cvt<<<(NW8 + 255) / 256, 256, 0, stream>>>(w1, w1bf, NW8);
  moe_cvt<<<(NW8 + 255) / 256, 256, 0, stream>>>(w2, w2bf, NW8);
  moe_cvt<<<(NW8 + 255) / 256, 256, 0, stream>>>(wp, wpbf, NW8);
  moe_gate<<<T_TOK / 8, 256, 0, stream>>>(x, gw, logits);
  moe_route1<<<T_TOK / 256, 256, 0, stream>>>(logits, noise, nw, top_idx, top_w,
                                              counts, usage);
  moe_setup<<<1, 1, 0, stream>>>(counts, usage, offsets, tile_exp, out_loss);
  moe_route2<<<T_TOK / 256, 256, 0, stream>>>(top_idx, top_w, offsets, fill,
                                              rowtok, rowwgt);
  moe_ffn1<<<dim3(MAX_TILES, HDIM / BN), 256, 0, stream>>>(xbf, w1bf, b1, w2bf, b2,
                                                           tile_exp, rowtok, zrow, hact);
  moe_ffn2<<<dim3(MAX_TILES, DDIM / BN), 256, 0, stream>>>(hact, wpbf, bp, tile_exp,
                                                           rowtok, rowwgt, out);
}
